// eHXBSfnHAaGpWFgNmODAIYNswTKSTBje_84980222919366
// MI455X (gfx1250) — matrix operand order repaired and checked by reading, not compiled, not run on hardware
//
#include <hip/hip_runtime.h>
#include <hip/hip_bf16.h>

#define B_ 2
#define C_ 512
#define N_ 4096
#define G_ 32
#define CPG (C_ / G_)                 // 16 channels per group
#define ATTN_SCALE 0.044194173824159216f  // 1/sqrt(512)
#define GN_EPS 1e-6f

typedef __attribute__((ext_vector_type(16))) _Float16 v16h;
typedef __attribute__((ext_vector_type(8)))  _Float16 v8h;
typedef __attribute__((ext_vector_type(8)))  float    v8f;

// ---------------------------------------------------------------------------
// WMMA fragment helpers (16-bit, contiguous k = 16h + s order chosen by this code, wave32)
//
// A (16xK=32): lane 0-15 holds row M=lane, K=0..15;
//              lane 16-31 holds row M=lane-16, K=16..31.
// Source must be row-major [M][K] (stride in halves).
__device__ __forceinline__ v16h load_a_frag(const _Float16* base, int row0,
                                            int col0, int stride, int lane) {
  int r = row0 + (lane & 15);
  int h = (lane >> 4) & 1;
  const _Float16* p = base + (long)r * stride + col0 + h * 16;
  return *(const v16h*)p;  // K = h*16 .. h*16+15
}

// B (K=32 x 16): lane 0-15 holds column N=lane, K=0..15;
//                lane 16-31 holds column N=lane-16, K=16..31.
// Source layout: element B[k][n] stored at base[n*stride + k]
// (n-major, k contiguous) -> one aligned 32-byte load per lane.
__device__ __forceinline__ v16h load_b_frag(const _Float16* base, int n0,
                                            int k0, int stride, int lane) {
  int n = n0 + (lane & 15);
  int h = (lane >> 4) & 1;
  return *(const v16h*)(base + (long)n * stride + k0 + h * 16);
}

__device__ __forceinline__ v8f wmma_f16(v16h a, v16h b, v8f c) {
  return __builtin_amdgcn_wmma_f32_16x16x32_f16(
      /*neg_a=*/false, a, /*neg_b=*/false, b,
      /*c_mod=*/(short)0, c, /*reuse_a=*/false, /*reuse_b=*/false);
}

// ---------------------------------------------------------------------------
// Kernel 1: fp32 -> f16 weight conversion for the four [C,C] matrices.
__global__ void wcvt_kernel(const float* __restrict__ wq, const float* __restrict__ wk,
                            const float* __restrict__ wv, const float* __restrict__ wo,
                            _Float16* __restrict__ q16, _Float16* __restrict__ k16,
                            _Float16* __restrict__ v16, _Float16* __restrict__ o16) {
  int i = blockIdx.x * blockDim.x + threadIdx.x;
  if (i < C_ * C_) {
    q16[i] = (_Float16)wq[i];
    k16[i] = (_Float16)wk[i];
    v16[i] = (_Float16)wv[i];
    o16[i] = (_Float16)wo[i];
  }
}

// ---------------------------------------------------------------------------
// Kernel 2: GroupNorm. One block per (batch, group); writes hT as [B][N][C] f16
// so it can feed WMMA A-fragments directly.
__global__ __launch_bounds__(256) void gnorm_kernel(const float* __restrict__ x,
                                                    const float* __restrict__ gw,
                                                    const float* __restrict__ gb,
                                                    _Float16* __restrict__ hT) {
  __shared__ float s_sum[256], s_sq[256];
  __shared__ float s_mean, s_rstd;
  int bg = blockIdx.x;  // 0 .. B*G-1
  int b = bg >> 5;
  int g = bg & 31;
  int c0 = g * CPG;
  const float* xg = x + ((long)b * C_ + c0) * N_;  // contiguous CPG*N_ floats
  const int cnt = CPG * N_;

  float sum = 0.f, sq = 0.f;
  for (int e = threadIdx.x; e < cnt; e += 256) {
    float v = xg[e];
    sum += v;
    sq += v * v;
  }
  s_sum[threadIdx.x] = sum;
  s_sq[threadIdx.x] = sq;
  __syncthreads();
  for (int s = 128; s > 0; s >>= 1) {
    if (threadIdx.x < s) {
      s_sum[threadIdx.x] += s_sum[threadIdx.x + s];
      s_sq[threadIdx.x] += s_sq[threadIdx.x + s];
    }
    __syncthreads();
  }
  if (threadIdx.x == 0) {
    float mean = s_sum[0] / (float)cnt;
    float var = s_sq[0] / (float)cnt - mean * mean;
    s_mean = mean;
    s_rstd = rsqrtf(var + GN_EPS);
  }
  __syncthreads();
  float mean = s_mean, rstd = s_rstd;
  for (int e = threadIdx.x; e < cnt; e += 256) {
    int cl = e >> 12;       // e / N_
    int n = e & (N_ - 1);   // e % N_
    int c = c0 + cl;
    float v = (xg[e] - mean) * rstd * gw[c] + gb[c];
    hT[((long)b * N_ + n) * C_ + c] = (_Float16)v;
  }
}

// ---------------------------------------------------------------------------
// Kernel 3: Q/K/V projections. One wave per 16x16 output tile; the A-fragment
// (hT tile) is reused for all three GEMMs. qT,kT stored [B][N][C]; V stored
// [B][C][N] (B-fragment-friendly for P@V^T).
__global__ __launch_bounds__(256) void qkv_kernel(const _Float16* __restrict__ hT,
                                                  const _Float16* __restrict__ wq,
                                                  const _Float16* __restrict__ wk,
                                                  const _Float16* __restrict__ wv,
                                                  const float* __restrict__ bq,
                                                  const float* __restrict__ bk,
                                                  const float* __restrict__ bv,
                                                  _Float16* __restrict__ qT,
                                                  _Float16* __restrict__ kT,
                                                  _Float16* __restrict__ vC) {
  int lane = threadIdx.x & 31;
  int gw = blockIdx.x * 8 + (threadIdx.x >> 5);  // global wave id
  int b = gw >> 13;                              // / (256*32)
  int rem = gw & 8191;
  int i0 = (rem >> 5) * 16;  // spatial tile
  int c0 = (rem & 31) * 16;  // output-channel tile
  const _Float16* hb = hT + (long)b * N_ * C_;

  v8f aq = {}, ak = {}, av = {};
  for (int k0 = 0; k0 < C_; k0 += 32) {
    v16h a = load_a_frag(hb, i0, k0, C_, lane);
    aq = wmma_f16(a, load_b_frag(wq, c0, k0, C_, lane), aq);
    ak = wmma_f16(a, load_b_frag(wk, c0, k0, C_, lane), ak);
    av = wmma_f16(a, load_b_frag(wv, c0, k0, C_, lane), av);
  }
  int cc = c0 + (lane & 15);
  float biq = bq[cc], bik = bk[cc], biv = bv[cc];
  int mofs = (lane & 16) >> 1;  // lanes 16-31 hold rows M=8..15
#pragma unroll
  for (int r = 0; r < 8; ++r) {
    int i = i0 + r + mofs;
    qT[((long)b * N_ + i) * C_ + cc] = (_Float16)(aq[r] + biq);
    kT[((long)b * N_ + i) * C_ + cc] = (_Float16)(ak[r] + bik);
    vC[((long)b * C_ + cc) * N_ + i] = (_Float16)(av[r] + biv);
  }
}

// ---------------------------------------------------------------------------
// Kernel 4: fused attention. One block (8 waves) per 16-query tile.
//   Phase 1: scores = scale * Q^T K  -> 128 KB LDS strip [16][4096] (f16)
//   Phase 2: block-wide softmax stats over the strip
//   Phase 3: out^T = P @ V^T, each wave owns 64 output channels
__global__ __launch_bounds__(256) void attn_kernel(const _Float16* __restrict__ qT,
                                                   const _Float16* __restrict__ kT,
                                                   const _Float16* __restrict__ vC,
                                                   _Float16* __restrict__ attnT) {
  __shared__ __align__(32) _Float16 P[16 * N_];  // 128 KB (CDNA5: 320KB/WGP)
  __shared__ float red[256];
  __shared__ float rowmax[16];
  __shared__ float rowsum[16];

  int lane = threadIdx.x & 31;
  int w = threadIdx.x >> 5;      // wave 0..7
  int b = blockIdx.x >> 8;       // / 256
  int i0 = (blockIdx.x & 255) * 16;
  const _Float16* qb = qT + (long)b * N_ * C_;
  const _Float16* kb = kT + (long)b * N_ * C_;
  const _Float16* vb = vC + (long)b * C_ * N_;

  // ---- Phase 1: this wave computes scores for keys [w*512, w*512+512) ----
  for (int jt = 0; jt < 32; ++jt) {
    int j0 = w * 512 + jt * 16;
    v8f sacc = {};
    for (int k0 = 0; k0 < C_; k0 += 32) {
      v16h a = load_a_frag(qb, i0, k0, C_, lane);
      v16h bb = load_b_frag(kb, j0, k0, C_, lane);
      sacc = wmma_f16(a, bb, sacc);
    }
    int mofs = (lane & 16) >> 1;
#pragma unroll
    for (int r = 0; r < 8; ++r)
      P[(r + mofs) * N_ + j0 + (lane & 15)] = (_Float16)(sacc[r] * ATTN_SCALE);
  }
  __syncthreads();

  // ---- Phase 2: softmax over keys (rows = 16 queries, 16 threads/row) ----
  int row = threadIdx.x >> 4;
  int seg = threadIdx.x & 15;
  float mx = -3.0e38f;
  for (int t = 0; t < 256; ++t)
    mx = fmaxf(mx, (float)P[row * N_ + seg * 256 + t]);
  red[threadIdx.x] = mx;
  __syncthreads();
  for (int s = 8; s > 0; s >>= 1) {
    if (seg < s) red[threadIdx.x] = fmaxf(red[threadIdx.x], red[threadIdx.x + s]);
    __syncthreads();
  }
  if (seg == 0) rowmax[row] = red[threadIdx.x];
  __syncthreads();
  float rmax = rowmax[row];
  float sum = 0.f;
  for (int t = 0; t < 256; ++t) {
    int idx = row * N_ + seg * 256 + t;
    float p = __expf((float)P[idx] - rmax);
    P[idx] = (_Float16)p;
    sum += p;
  }
  red[threadIdx.x] = sum;
  __syncthreads();
  for (int s = 8; s > 0; s >>= 1) {
    if (seg < s) red[threadIdx.x] += red[threadIdx.x + s];
    __syncthreads();
  }
  if (seg == 0) rowsum[row] = red[threadIdx.x];
  __syncthreads();

  // ---- Phase 3: P @ V^T, this wave handles channels [w*64, w*64+64) ----
  v8f oacc[4] = {};
  for (int jc = 0; jc < N_; jc += 32) {
    v16h a = load_a_frag(P, 0, jc, N_, lane);  // A from LDS probability strip
#pragma unroll
    for (int t = 0; t < 4; ++t) {
      int c0 = w * 64 + t * 16;
      v16h bb = load_b_frag(vb, c0, jc, N_, lane);
      oacc[t] = wmma_f16(a, bb, oacc[t]);
    }
  }
  int mofs = (lane & 16) >> 1;
#pragma unroll
  for (int t = 0; t < 4; ++t) {
    int cc = w * 64 + t * 16 + (lane & 15);
#pragma unroll
    for (int r = 0; r < 8; ++r) {
      int m = r + mofs;
      float o = oacc[t][r] / rowsum[m];
      attnT[((long)b * N_ + i0 + m) * C_ + cc] = (_Float16)o;
    }
  }
}

// ---------------------------------------------------------------------------
// Kernel 5: output projection + bias + residual, fp32 result.
__global__ __launch_bounds__(256) void oproj_kernel(const _Float16* __restrict__ attnT,
                                                    const _Float16* __restrict__ wo,
                                                    const float* __restrict__ bo,
                                                    const float* __restrict__ x,
                                                    float* __restrict__ out) {
  int lane = threadIdx.x & 31;
  int gw = blockIdx.x * 8 + (threadIdx.x >> 5);
  int b = gw >> 13;
  int rem = gw & 8191;
  int i0 = (rem >> 5) * 16;
  int o0 = (rem & 31) * 16;
  const _Float16* ab = attnT + (long)b * N_ * C_;

  v8f acc = {};
  for (int k0 = 0; k0 < C_; k0 += 32) {
    v16h a = load_a_frag(ab, i0, k0, C_, lane);
    v16h bb = load_b_frag(wo, o0, k0, C_, lane);
    acc = wmma_f16(a, bb, acc);
  }
  int oo = o0 + (lane & 15);
  float bias = bo[oo];
  int mofs = (lane & 16) >> 1;
#pragma unroll
  for (int r = 0; r < 8; ++r) {
    long idx = ((long)b * C_ + oo) * N_ + i0 + r + mofs;
    out[idx] = acc[r] + bias + x[idx];
  }
}

// ---------------------------------------------------------------------------
extern "C" void kernel_launch(void* const* d_in, const int* in_sizes, int n_in,
                              void* d_out, int out_size, void* d_ws, size_t ws_size,
                              hipStream_t stream) {
  const float* x  = (const float*)d_in[0];
  const float* nw = (const float*)d_in[1];
  const float* nb = (const float*)d_in[2];
  const float* wq = (const float*)d_in[3];
  const float* bq = (const float*)d_in[4];
  const float* wk = (const float*)d_in[5];
  const float* bk = (const float*)d_in[6];
  const float* wv = (const float*)d_in[7];
  const float* bv = (const float*)d_in[8];
  const float* wo = (const float*)d_in[9];
  const float* bo = (const float*)d_in[10];
  float* out = (float*)d_out;

  char* ws = (char*)d_ws;
  const size_t SZ = (size_t)B_ * N_ * C_ * sizeof(_Float16);  // 8 MB each
  _Float16* hT    = (_Float16*)(ws);            // [B][N][C]
  _Float16* qT    = (_Float16*)(ws + SZ);       // [B][N][C]
  _Float16* kT    = (_Float16*)(ws + 2 * SZ);   // [B][N][C]
  _Float16* vC    = (_Float16*)(ws + 3 * SZ);   // [B][C][N]
  _Float16* attnT = (_Float16*)(ws + 4 * SZ);   // [B][N][C]
  _Float16* wq16  = (_Float16*)(ws + 5 * SZ);   // 4x [C][C] f16
  _Float16* wk16  = wq16 + C_ * C_;
  _Float16* wv16  = wk16 + C_ * C_;
  _Float16* wo16  = wv16 + C_ * C_;

  wcvt_kernel<<<(C_ * C_ + 255) / 256, 256, 0, stream>>>(wq, wk, wv, wo,
                                                         wq16, wk16, wv16, wo16);
  gnorm_kernel<<<B_ * G_, 256, 0, stream>>>(x, nw, nb, hT);
  qkv_kernel<<<(B_ * (N_ / 16) * (C_ / 16)) / 8, 256, 0, stream>>>(
      hT, wq16, wk16, wv16, bq, bk, bv, qT, kT, vC);
  attn_kernel<<<B_ * (N_ / 16), 256, 0, stream>>>(qT, kT, vC, attnT);
  oproj_kernel<<<(B_ * (N_ / 16) * (C_ / 16)) / 8, 256, 0, stream>>>(
      attnT, wo16, bo, x, out);
}
